// MultiHeadSelfAttention_73710228734744
// MI455X (gfx1250) — matrix operand order repaired and checked by reading, not compiled, not run on hardware
//
#include <hip/hip_runtime.h>
#include <hip/hip_bf16.h>

// ---------------------------------------------------------------------------
// MHA forward for gfx1250 (MI455X): f16 WMMA (16x16x32), f32 accumulate.
//   kernel 1: C = A * W^T   — A tiles via global_load_async_to_lds (ASYNCcnt),
//                             W tiles via TDM tensor_load_to_lds (TENSORcnt),
//                             double-buffered in LDS, f32->f16 at frag build.
//   kernel 2: RoPE in-place on Q,K
//   kernel 3: flash-style causal attention, 1 wave per 16-row Q tile
// ---------------------------------------------------------------------------

typedef __attribute__((ext_vector_type(16))) _Float16     v16h;
typedef __attribute__((ext_vector_type(8)))  float        v8f;
typedef __attribute__((ext_vector_type(4)))  float        v4f;
typedef __attribute__((ext_vector_type(4)))  unsigned int v4u;
typedef __attribute__((ext_vector_type(8)))  int          v8i;
typedef __attribute__((ext_vector_type(4)))  int          v4i;

union HFrag { _Float16 h[16]; v16h v; };

#define D_MODEL   1024
#define NUM_HEADS 16
#define D_HEAD    64
#define LDF       36      // padded f32 LDS row stride (32 data + 4 pad dwords)

// low 32 bits of a flat LDS pointer == LDS byte address (aperture rule)
__device__ __forceinline__ unsigned lds_lo32(const void* p) {
  return (unsigned)(size_t)p;
}

// ---------------------------------------------------------------------------
// GEMM: C[M,N] = A[M,K] * W[N,K]^T  (f32 in/out, f16 WMMA inside)
// Block 256 threads (8 waves), tile 128x128, K-step 32, double-buffered LDS.
// ---------------------------------------------------------------------------
__global__ __launch_bounds__(256) void gemm_xwt_f16_wmma(
    const float* __restrict__ A, const float* __restrict__ W,
    float* __restrict__ C, int M, int N, int K)
{
  __shared__ float As[2][128 * LDF];   // activations, filled by async-to-LDS
  __shared__ float Ws[2][128 * LDF];   // weights, filled by TDM

  const int tid  = threadIdx.x;
  const int lane = tid & 31;
  const int wave = tid >> 5;
  const int wm   = wave & 3;           // 4 waves along M
  const int wn   = wave >> 2;          // 2 waves along N
  const int half = lane >> 4;
  const int l15  = lane & 15;
  const int blockM = blockIdx.y * 128;
  const int blockN = blockIdx.x * 128;

  v8f vzero = {};
  v8f acc[2][4];
#pragma unroll
  for (int i = 0; i < 2; ++i)
#pragma unroll
    for (int j = 0; j < 4; ++j) acc[i][j] = vzero;

  // per-thread async-copy mapping: row = tid/2, 16-f32 chunk = (tid&1)*16
  const int ldrow = tid >> 1;
  const int ldk   = (tid & 1) * 16;
  const float* aBase = A + (size_t)(blockM + ldrow) * K + ldk;

  // issue 4 x b128 async copies of this thread's slice of the A tile
  auto issueA = [&](int k0, int buf) {
    unsigned lbase = lds_lo32(&As[buf][ldrow * LDF + ldk]);
    const float* g = aBase + k0;
#pragma unroll
    for (int e = 0; e < 16; e += 4) {
      unsigned l = lbase + e * 4;
      const float* ga = g + e;
      asm volatile("global_load_async_to_lds_b128 %0, %1, off"
                   :: "v"(l), "v"(ga) : "memory");
    }
  };

  // wave 0 drives the Tensor Data Mover for the 128xK-slice of W
  auto issueW = [&](int k0, int buf) {
    if (tid < 32) {
      unsigned lds = lds_lo32(&Ws[buf][0]);
      unsigned long long ga =
          (unsigned long long)(const void*)(W + (size_t)blockN * K + k0);
      v4u g0; v8i g1; v4i gz = {0, 0, 0, 0};
      g0[0] = 1u;                                       // count=1, user mode
      g0[1] = lds;                                      // lds_addr
      g0[2] = (unsigned)(ga & 0xFFFFFFFFu);             // global_addr[31:0]
      g0[3] = (unsigned)((ga >> 32) & 0x1FFFFFFu)       // global_addr[56:32]
            | (2u << 30);                               // type = 2 (image)
      g1[0] = (int)((2u << 16)                          // data_size = 4B
            | (1u << 20)                                // pad_enable
            | (4u << 22)                                // pad every 32 dwords
            | (3u << 25));                              // pad amount 4 dwords
      g1[1] = (int)(((unsigned)K & 0xFFFFu) << 16);     // tensor_dim0 lo16
      g1[2] = (int)((((unsigned)K >> 16) & 0xFFFFu)     // tensor_dim0 hi16
            | (((unsigned)N & 0xFFFFu) << 16));         // tensor_dim1 lo16
      g1[3] = (int)((((unsigned)N >> 16) & 0xFFFFu)     // tensor_dim1 hi16
            | (32u << 16));                             // tile_dim0 = 32
      g1[4] = (int)(128u);                              // tile_dim1=128, dim2=0
      g1[5] = (int)((unsigned)K);                       // dim0_stride lo32
      g1[6] = 0;                                        // stride hi, d1stride lo
      g1[7] = 0;
#if __clang_major__ >= 23
      v8i gz8 = {0, 0, 0, 0, 0, 0, 0, 0};
      __builtin_amdgcn_tensor_load_to_lds(g0, g1, gz, gz, gz8, 0);
#else
      __builtin_amdgcn_tensor_load_to_lds(g0, g1, gz, gz, 0);
#endif
    }
  };

  const int nk = K / 32;
  issueW(0, 0);
  issueA(0, 0);

  for (int kt = 0; kt < nk; ++kt) {
    const int cur = kt & 1;
    if (kt + 1 < nk) {                       // keep next tile in flight
      issueW((kt + 1) * 32, cur ^ 1);
      issueA((kt + 1) * 32, cur ^ 1);
      if (tid < 32) __builtin_amdgcn_s_wait_tensorcnt(1);
      asm volatile("s_wait_asynccnt 0x4" ::: "memory");
    } else {
      if (tid < 32) __builtin_amdgcn_s_wait_tensorcnt(0);
      asm volatile("s_wait_asynccnt 0x0" ::: "memory");
    }
    __syncthreads();

    // A fragment (16x32 f16): lanes 0-15 row m hold K 0..7 & 16..23,
    //                         lanes 16-31 same row hold K 8..15 & 24..31.
    HFrag aF[2], bF[4];
    const int aKlo = half * 8;
    const int bKlo = half * 8;
#pragma unroll
    for (int i = 0; i < 2; ++i) {
      const float* p = &As[cur][(wm * 32 + i * 16 + l15) * LDF];
#pragma unroll
      for (int e = 0; e < 8; ++e) {
        aF[i].h[e]     = (_Float16)p[aKlo + e];
        aF[i].h[8 + e] = (_Float16)p[aKlo + 16 + e];
      }
    }
    // B fragment (32x16 f16): column n = lane&15; lanes 0-15 K 0..7 & 16..23,
    //                         lanes 16-31 K 8..15 & 24..31 (same as A).
#pragma unroll
    for (int j = 0; j < 4; ++j) {
      const float* p = &Ws[cur][(wn * 64 + j * 16 + l15) * LDF + bKlo];
#pragma unroll
      for (int e = 0; e < 8; ++e) { bF[j].h[e] = (_Float16)p[e]; bF[j].h[8 + e] = (_Float16)p[16 + e]; }
    }

#pragma unroll
    for (int i = 0; i < 2; ++i)
#pragma unroll
      for (int j = 0; j < 4; ++j)
        acc[i][j] = __builtin_amdgcn_wmma_f32_16x16x32_f16(
            false, aF[i].v, false, bF[j].v, (short)0, acc[i][j], false, false);
    __syncthreads();                          // buffer safe to refill
  }

  // C/D layout: VGPR r -> row r (lanes 0-15) / row r+8 (lanes 16-31), col=l15
#pragma unroll
  for (int i = 0; i < 2; ++i) {
    const int mBase = blockM + wm * 32 + i * 16 + half * 8;
#pragma unroll
    for (int j = 0; j < 4; ++j) {
      const int n = blockN + wn * 64 + j * 16 + l15;
#pragma unroll
      for (int r = 0; r < 8; ++r)
        C[(size_t)(mBase + r) * N + n] = acc[i][j][r];
    }
  }
}

// ---------------------------------------------------------------------------
// RoPE in-place on Q and K: one thread per (b,s,h,pair)
// ---------------------------------------------------------------------------
__global__ void rope_inplace(float* __restrict__ q, float* __restrict__ kk,
                             int B, int S)
{
  const int pairs = D_HEAD / 2;                       // 32
  const int idx = blockIdx.x * blockDim.x + threadIdx.x;
  const int total = B * S * NUM_HEADS * pairs;
  if (idx >= total) return;
  const int p = idx & (pairs - 1);
  int t = idx / pairs;
  const int h = t & (NUM_HEADS - 1);
  t >>= 4;                                            // / NUM_HEADS
  const int s = t % S;
  const int b = t / S;

  const float ang = (float)s * __powf(10000.0f, -(float)(2 * p) / (float)D_HEAD);
  float sn, cs;
  __sincosf(ang, &sn, &cs);

  const size_t base = ((size_t)(b * S + s) * D_MODEL) + h * D_HEAD + 2 * p;
  const float qe = q[base], qo = q[base + 1];
  q[base]     = qe * cs - qo * sn;
  q[base + 1] = qe * sn + qo * cs;
  const float ke = kk[base], ko = kk[base + 1];
  kk[base]     = ke * cs - ko * sn;
  kk[base + 1] = ke * sn + ko * cs;
}

// ---------------------------------------------------------------------------
// Flash-style causal attention. One wave per (qtile=16 rows, head, batch).
// Per 32-key step: 4 WMMAs for S = Q*K^T, online softmax (shfl_xor half-wave
// reductions), P round-trips through LDS (C-layout -> A-layout), V staged
// f32->f16 in LDS, 4 WMMAs for O += P*V.  1/sqrt(Dh) folded into Q.
// ---------------------------------------------------------------------------
__global__ __launch_bounds__(32) void attn_fwd_wmma(
    const float* __restrict__ q, const float* __restrict__ k,
    const float* __restrict__ v, float* __restrict__ o,
    int B, int S)
{
  __shared__ _Float16 Pl[16 * 40];   // 16 x 32 probs, padded
  __shared__ _Float16 Vs[32 * 72];   // 32 keys x 64 dh, padded

  const int lane = threadIdx.x;
  const int half = lane >> 4;
  const int l15  = lane & 15;
  const int qt = blockIdx.x, h = blockIdx.y, b = blockIdx.z;
  const int D = D_MODEL;
  const int qBase = qt * 16;
  const float scale = 0.125f;        // 1/sqrt(64)

  HFrag aQ[2];
  {
    const float* qp = q + ((size_t)(b * S + qBase + l15) * D) + h * D_HEAD;
    const int kLo = half * 8;
#pragma unroll
    for (int c = 0; c < 2; ++c)
#pragma unroll
      for (int e = 0; e < 8; ++e) {
        aQ[c].h[e]     = (_Float16)(qp[c * 32 + kLo + e] * scale);
        aQ[c].h[8 + e] = (_Float16)(qp[c * 32 + kLo + 16 + e] * scale);
      }
  }

  float mrow[8], lrow[8];
  v8f vzero = {};
  v8f accO[4];
#pragma unroll
  for (int r = 0; r < 8; ++r) { mrow[r] = -1e30f; lrow[r] = 0.f; }
#pragma unroll
  for (int j = 0; j < 4; ++j) accO[j] = vzero;

  for (int kb = 0; kb <= qBase; kb += 32) {
    HFrag bK[2][2];
#pragma unroll
    for (int t = 0; t < 2; ++t) {
      int key = kb + t * 16 + l15; if (key >= S) key = S - 1;
      const float* kp = k + ((size_t)(b * S + key) * D) + h * D_HEAD;
      const int kB = half * 8;
#pragma unroll
      for (int c = 0; c < 2; ++c)
#pragma unroll
        for (int e = 0; e < 8; ++e) {
          bK[t][c].h[e]     = (_Float16)kp[c * 32 + kB + e];
          bK[t][c].h[8 + e] = (_Float16)kp[c * 32 + kB + 16 + e];
        }
    }

    v8f st0 = vzero, st1 = vzero;
#pragma unroll
    for (int c = 0; c < 2; ++c) {
      st0 = __builtin_amdgcn_wmma_f32_16x16x32_f16(false, aQ[c].v, false, bK[0][c].v, (short)0, st0, false, false);
      st1 = __builtin_amdgcn_wmma_f32_16x16x32_f16(false, aQ[c].v, false, bK[1][c].v, (short)0, st1, false, false);
    }

    {
      int key = kb + lane; if (key >= S) key = S - 1;
      const float* vp = v + ((size_t)(b * S + key) * D) + h * D_HEAD;
      _Float16* dst = &Vs[lane * 72];
#pragma unroll
      for (int e = 0; e < 64; e += 4) {
        v4f x = *(const v4f*)(vp + e);
        dst[e]     = (_Float16)x.x; dst[e + 1] = (_Float16)x.y;
        dst[e + 2] = (_Float16)x.z; dst[e + 3] = (_Float16)x.w;
      }
    }

#pragma unroll
    for (int r = 0; r < 8; ++r) {
      const int m = r + half * 8;
      const int limit = qBase + m;
      float s0 = st0[r], s1 = st1[r];
      s0 = ((kb + l15)      <= limit) ? s0 : -1e30f;
      s1 = ((kb + 16 + l15) <= limit) ? s1 : -1e30f;
      float tmax = fmaxf(s0, s1);
#pragma unroll
      for (int off = 1; off < 16; off <<= 1)
        tmax = fmaxf(tmax, __shfl_xor(tmax, off, 32));
      const float mnew = fmaxf(mrow[r], tmax);
      const float corr = __expf(mrow[r] - mnew);
      const float p0 = __expf(s0 - mnew);
      const float p1 = __expf(s1 - mnew);
      float rs = p0 + p1;
#pragma unroll
      for (int off = 1; off < 16; off <<= 1)
        rs += __shfl_xor(rs, off, 32);
      lrow[r] = lrow[r] * corr + rs;
      mrow[r] = mnew;
#pragma unroll
      for (int j = 0; j < 4; ++j) accO[j][r] *= corr;
      Pl[m * 40 + l15]      = (_Float16)p0;
      Pl[m * 40 + 16 + l15] = (_Float16)p1;
    }
    __syncthreads();

    HFrag aP;
    {
      const int kLo = half * 8;
      const _Float16* p = &Pl[l15 * 40];
#pragma unroll
      for (int e = 0; e < 8; ++e) { aP.h[e] = p[kLo + e]; aP.h[8 + e] = p[kLo + 16 + e]; }
    }
#pragma unroll
    for (int j = 0; j < 4; ++j) {
      HFrag bV;
      const int kB = half * 8;
#pragma unroll
      for (int e = 0; e < 8; ++e) { bV.h[e] = Vs[(kB + e) * 72 + j * 16 + l15]; bV.h[8 + e] = Vs[(kB + 16 + e) * 72 + j * 16 + l15]; }
      accO[j] = __builtin_amdgcn_wmma_f32_16x16x32_f16(false, aP.v, false, bV.v, (short)0, accO[j], false, false);
    }
    __syncthreads();
  }

#pragma unroll
  for (int j = 0; j < 4; ++j) {
    const int n = h * D_HEAD + j * 16 + l15;
#pragma unroll
    for (int r = 0; r < 8; ++r) {
      const int m = r + half * 8;
      o[((size_t)(b * S + qBase + m) * D) + n] = accO[j][r] / lrow[r];
    }
  }
}

// ---------------------------------------------------------------------------
extern "C" void kernel_launch(void* const* d_in, const int* in_sizes, int n_in,
                              void* d_out, int out_size, void* d_ws, size_t ws_size,
                              hipStream_t stream) {
  const float* x  = (const float*)d_in[0];
  const float* wq = (const float*)d_in[1];
  const float* wk = (const float*)d_in[2];
  const float* wv = (const float*)d_in[3];
  const float* wo = (const float*)d_in[4];
  float* out = (float*)d_out;
  (void)n_in; (void)out_size; (void)ws_size;

  const int S = 2048, D = D_MODEL;
  const int B = in_sizes[0] / (S * D);   // = 2
  const int M = B * S;                   // = 4096

  const size_t elems = (size_t)M * D;    // 4 buffers * 16 MB = 64 MB scratch
  float* qb = (float*)d_ws;
  float* kb = qb + elems;
  float* vb = kb + elems;
  float* ab = vb + elems;

  dim3 gblk(256);
  dim3 ggrid(D / 128, M / 128);
  gemm_xwt_f16_wmma<<<ggrid, gblk, 0, stream>>>(x, wq, qb, M, D, D);
  gemm_xwt_f16_wmma<<<ggrid, gblk, 0, stream>>>(x, wk, kb, M, D, D);
  gemm_xwt_f16_wmma<<<ggrid, gblk, 0, stream>>>(x, wv, vb, M, D, D);

  const int total = B * S * NUM_HEADS * (D_HEAD / 2);
  rope_inplace<<<(total + 255) / 256, 256, 0, stream>>>(qb, kb, B, S);

  dim3 agrid(S / 16, NUM_HEADS, B);
  attn_fwd_wmma<<<agrid, dim3(32), 0, stream>>>(qb, kb, vb, ab, B, S);

  gemm_xwt_f16_wmma<<<ggrid, gblk, 0, stream>>>(ab, wo, out, M, D, D);
}
